// ScaledDotProductAttentionLayer_7825430413614
// MI455X (gfx1250) — matrix operand order repaired and checked by reading, not compiled, not run on hardware
//
#include <hip/hip_runtime.h>

typedef __attribute__((ext_vector_type(16))) _Float16 v16h;
typedef __attribute__((ext_vector_type(8)))  float    v8f;
typedef __attribute__((ext_vector_type(4)))  float    v4f;

constexpr int B = 2, H = 16, S = 2048, D = 128;
constexpr int QTILE = 128;   // q rows per block (8 waves x 16 rows)
constexpr int KTILE = 32;    // kv rows per iteration
constexpr int NT    = S / KTILE;
constexpr int KSTR  = 136;   // smK row stride in halfs ([kv][d], padded vs 128)
constexpr int VSTR  = 40;    // smV row stride in halfs ([d][kv], padded vs 32)
constexpr int PSTR  = 40;    // smP row stride in halfs (per-wave 16x32 bounce)

struct Stage { v4f k[4]; v4f v[4]; };   // 32 VGPRs of in-flight tile data

__device__ __forceinline__ v16h ld16(const _Float16* p) {
  v16h r;
#pragma unroll
  for (int i = 0; i < 16; ++i) r[i] = p[i];
  return r;
}
__device__ __forceinline__ v16h ld8x2(const _Float16* p0, const _Float16* p1) {
  v16h r;
#pragma unroll
  for (int i = 0; i < 8; ++i) { r[i] = p0[i]; r[8 + i] = p1[i]; }
  return r;
}
// reductions across the 16 lanes of a half-wave (wave32)
__device__ __forceinline__ float rmax16(float v) {
#pragma unroll
  for (int m = 1; m < 16; m <<= 1) v = fmaxf(v, __shfl_xor(v, m, 32));
  return v;
}
__device__ __forceinline__ float rsum16(float v) {
#pragma unroll
  for (int m = 1; m < 16; m <<= 1) v += __shfl_xor(v, m, 32);
  return v;
}

__device__ __forceinline__ void load_stage(const float* __restrict__ gk,
                                           const float* __restrict__ gv,
                                           int tid, Stage& st) {
#pragma unroll
  for (int j = 0; j < 4; ++j) {
    int idx = tid * 4 + 1024 * j;          // 0..4095 over the 32x128 tile
    st.k[j] = *(const v4f*)(gk + idx);
    st.v[j] = *(const v4f*)(gv + idx);
  }
}
__device__ __forceinline__ void store_stage(_Float16* bK, _Float16* bV,
                                            int tid, const Stage& st) {
#pragma unroll
  for (int j = 0; j < 4; ++j) {
    int idx = tid * 4 + 1024 * j;
    int r = idx >> 7, c = idx & 127;
#pragma unroll
    for (int u = 0; u < 4; ++u) {
      bK[r * KSTR + c + u]   = (_Float16)st.k[j][u];   // row-major K
      bV[(c + u) * VSTR + r] = (_Float16)st.v[j][u];   // transposed V
    }
  }
}

__global__ __launch_bounds__(256, 1) void fa_fwd_kernel(
    const float* __restrict__ Q, const float* __restrict__ K,
    const float* __restrict__ V, const int* __restrict__ M,
    float* __restrict__ O) {
  __shared__ alignas(32) _Float16 smK[2][KTILE * KSTR];  // double-buffered
  __shared__ alignas(32) _Float16 smV[2][D * VSTR];
  __shared__ alignas(32) _Float16 smP[8][16 * PSTR];     // per-wave C->A bounce

  const int tid  = threadIdx.x;
  const int wv   = tid >> 5;
  const int lane = tid & 31;
  const int ln   = lane & 15;   // column (B/C frags) or row (A frags)
  const int g    = lane >> 4;   // half-wave select

  const int bh    = blockIdx.y;                 // fused b*H + h
  const int b     = bh / H;
  const int qbase = blockIdx.x * QTILE + wv * 16;

  const size_t headoff = (size_t)bh * S * D;
  const float* gq = Q + headoff + (size_t)(qbase + ln) * D;
  const float* gk0 = K + headoff;
  const float* gv0 = V + headoff;

  // Q A-fragments: 16x32 f16 per 32-wide d-chunk.
  // A layout: lane holds row m = lane%16; elems 0..7 = K g*8..+7,
  // elems 8..15 = K 16+g*8..+7 (within the 32-wide chunk).
  v16h qf[4];
#pragma unroll
  for (int c = 0; c < 4; ++c) {
    v16h a;
#pragma unroll
    for (int j = 0; j < 8; ++j) {
      a[j]     = (_Float16)gq[c * 32 + g * 8 + j];
      a[8 + j] = (_Float16)gq[c * 32 + 16 + g * 8 + j];
    }
    qf[c] = a;
  }

  v8f zero;
#pragma unroll
  for (int i = 0; i < 8; ++i) zero[i] = 0.0f;

  v8f acc[8];              // O tile 16x128 f32: 8 N-tiles of 16 cols
#pragma unroll
  for (int t = 0; t < 8; ++t) acc[t] = zero;
  float mrow[8], lrow[8];  // per-lane row stats (row = r + 8*g)
#pragma unroll
  for (int r = 0; r < 8; ++r) { mrow[r] = -3.0e38f; lrow[r] = 0.0f; }

  const float scale = 0.088388347648318447f;  // 1/sqrt(128)
  const int* mbase = M + (size_t)b * S * S;

  // ---- prologue: fill buffer 0 with tile 0 ----------------------------
  Stage st;
  load_stage(gk0, gv0, tid, st);
  store_stage(smK[0], smV[0], tid, st);

  for (int kt = 0; kt < NT; ++kt) {
    __syncthreads();  // buf[kt&1] ready; prior reads of buf[(kt+1)&1] done
    const int cur = kt & 1;
    const int k0  = kt * KTILE;

    // Issue next tile's global loads first: they drain under the WMMA work.
    if (kt + 1 < NT)
      load_stage(gk0 + (size_t)(k0 + KTILE) * D,
                 gv0 + (size_t)(k0 + KTILE) * D, tid, st);

    const _Float16* bK = smK[cur];
    const _Float16* bV = smV[cur];

    // ---- scores S = Q K^T : 16 x 32 as two 16-col C tiles --------------
    v8f sc[2];
#pragma unroll
    for (int nt = 0; nt < 2; ++nt) {
      v16h kf[4];
#pragma unroll
      for (int dc = 0; dc < 4; ++dc)   // batch the 8 ds_load_b128 into a clause
        kf[dc] = ld8x2(&bK[(nt * 16 + ln) * KSTR + dc * 32 + g * 8],
                       &bK[(nt * 16 + ln) * KSTR + dc * 32 + 16 + g * 8]);
      v8f c = zero;
#pragma unroll
      for (int dc = 0; dc < 4; ++dc)
        c = __builtin_amdgcn_wmma_f32_16x16x32_f16(false, qf[dc], false, kf[dc],
                                                   (short)0, c, false, false);
      sc[nt] = c;
    }

    // ---- scale + additive mask ----------------------------------------
#pragma unroll
    for (int nt = 0; nt < 2; ++nt) {
#pragma unroll
      for (int r = 0; r < 8; ++r) {
        int qr = qbase + r + 8 * g;
        int kc = k0 + nt * 16 + ln;
        float mk = (float)mbase[(size_t)qr * S + kc];
        sc[nt][r] = sc[nt][r] * scale - 1.0e9f * mk;
      }
    }

    // ---- online softmax (per row, across 16 lanes + both N tiles) -----
    float pscale[8];
#pragma unroll
    for (int r = 0; r < 8; ++r) {
      float rowm = rmax16(fmaxf(sc[0][r], sc[1][r]));
      float mnew = fmaxf(mrow[r], rowm);
      float corr = __expf(mrow[r] - mnew);
      sc[0][r] = __expf(sc[0][r] - mnew);
      sc[1][r] = __expf(sc[1][r] - mnew);
      lrow[r] = lrow[r] * corr + rsum16(sc[0][r] + sc[1][r]);
      mrow[r] = mnew;
      pscale[r] = corr;
    }

    // rescale existing accumulators
#pragma unroll
    for (int t = 0; t < 8; ++t)
#pragma unroll
      for (int r = 0; r < 8; ++r) acc[t][r] *= pscale[r];

    // ---- P: C-layout -> A-layout via per-wave LDS bounce --------------
    _Float16* pw = smP[wv];
#pragma unroll
    for (int nt = 0; nt < 2; ++nt)
#pragma unroll
      for (int r = 0; r < 8; ++r)
        pw[(r + 8 * g) * PSTR + nt * 16 + ln] = (_Float16)sc[nt][r];
    // same-wave DS ops are in-order; loads below see the stores above
    v16h pf = ld8x2(&pw[ln * PSTR + g * 8], &pw[ln * PSTR + 16 + g * 8]);

    // ---- O += P V : 8 WMMAs over the 8 d-tiles, fragments batched x4 --
#pragma unroll
    for (int h2 = 0; h2 < 2; ++h2) {
      v16h vf[4];
#pragma unroll
      for (int t4 = 0; t4 < 4; ++t4) {
        int t = h2 * 4 + t4;
        vf[t4] = ld8x2(&bV[(t * 16 + ln) * VSTR + g * 8],
                       &bV[(t * 16 + ln) * VSTR + 16 + g * 8]);
      }
#pragma unroll
      for (int t4 = 0; t4 < 4; ++t4) {
        int t = h2 * 4 + t4;
        acc[t] = __builtin_amdgcn_wmma_f32_16x16x32_f16(false, pf, false, vf[t4],
                                                        (short)0, acc[t], false, false);
      }
    }

    // ---- drain staged loads into the other LDS buffer -----------------
    if (kt + 1 < NT)
      store_stage(smK[cur ^ 1], smV[cur ^ 1], tid, st);
  }

  // ---- epilogue: O / l ------------------------------------------------
  float* go = O + headoff + (size_t)qbase * D;
#pragma unroll
  for (int r = 0; r < 8; ++r) {
    float inv = 1.0f / lrow[r];
    float* row = go + (size_t)(r + 8 * g) * D;
#pragma unroll
    for (int t = 0; t < 8; ++t)
      row[t * 16 + ln] = acc[t][r] * inv;
  }
}

extern "C" void kernel_launch(void* const* d_in, const int* in_sizes, int n_in,
                              void* d_out, int out_size, void* d_ws, size_t ws_size,
                              hipStream_t stream) {
  // setup_inputs order: queries, keys, values, d_k, mask
  const float* Qp = (const float*)d_in[0];
  const float* Kp = (const float*)d_in[1];
  const float* Vp = (const float*)d_in[2];
  const int*   Mp = (const int*)d_in[4];   // d_in[3] = d_k (128, hardcoded)
  float* Op = (float*)d_out;

  dim3 grid(S / QTILE, B * H);
  fa_fwd_kernel<<<grid, 256, 0, stream>>>(Qp, Kp, Vp, Mp, Op);
}
